// LSTM_360
// MI455X (gfx1250) — compile-verified, matrix operand order checked by reading, not run on hardware
//
#include <hip/hip_runtime.h>
#include <hip/hip_bf16.h>

// ---------------------------------------------------------------------------
// LSTM forward for MI455X (gfx1250), bf16 WMMA with fp32 accumulate.
// SEQ=512, BATCH=128, IN=256, HID=1024.
//
//  * wcat = [Whh | Wxh] bf16 [4096][1280] (10 MB, L2-resident).
//  * One kernel per timestep. Blocks 0..63: gates GEMM (each block owns one
//    16-wide hidden column across all 4 gates; wave w owns batch rows
//    [16w,16w+16)) + fused cell update. K loop: 40 x 32 (h_{t-1} then x_t),
//    B-tiles double-buffered through LDS (one barrier per k-step).
//    Blocks 64..67: output projection for h_{t-1} (h double-buffered).
//  * Final log_softmax over [65536][256] rows.
// ---------------------------------------------------------------------------

typedef __attribute__((ext_vector_type(16))) __bf16 v16bf;
typedef __attribute__((ext_vector_type(8)))  float  v8f;

#define SEQ   512
#define BATCH 128
#define IND   256
#define HID   1024
#define G4    4096
#define KTOT  1280   // HID + IND
#define NKS   (KTOT / 32)

union FragU { uint4 q[2]; v16bf v; };

__device__ __forceinline__ unsigned short f2bf(float f) {
    unsigned u = __float_as_uint(f);
    unsigned r = (u + 0x7FFFu + ((u >> 16) & 1u)) >> 16;   // RNE
    return (unsigned short)r;
}

__device__ __forceinline__ v8f splat8(float b) {
    v8f v;
#pragma unroll
    for (int i = 0; i < 8; ++i) v[i] = b;
    return v;
}

// ------------------------------- prep kernels ------------------------------

__global__ void k_cast_wcat(const float* __restrict__ Whh,
                            const float* __restrict__ Wxh,
                            unsigned short* __restrict__ wcat) {
    int idx = blockIdx.x * 256 + threadIdx.x;          // G4*KTOT
    if (idx >= G4 * KTOT) return;
    int g = idx / KTOT, k = idx % KTOT;
    float v = (k < HID) ? Whh[(size_t)g * HID + k]
                        : Wxh[(size_t)g * IND + (k - HID)];
    wcat[idx] = f2bf(v);
}

__global__ void k_cast_f32_bf16(const float* __restrict__ src,
                                unsigned short* __restrict__ dst, int n) {
    int idx = blockIdx.x * 256 + threadIdx.x;
    if (idx < n) dst[idx] = f2bf(src[idx]);
}

__global__ void k_bias_cat(const float* __restrict__ bxh,
                           const float* __restrict__ bhh,
                           float* __restrict__ biascat) {
    int idx = blockIdx.x * 256 + threadIdx.x;
    if (idx < G4) biascat[idx] = bxh[idx] + bhh[idx];
}

__global__ void k_init_state(float* __restrict__ cst,
                             unsigned short* __restrict__ hbuf) {
    int idx = blockIdx.x * 256 + threadIdx.x;
    if (idx < 2 * BATCH * HID) hbuf[idx] = 0;          // both h buffers
    if (idx < BATCH * HID)     cst[idx]  = 0.0f;
}

// ------------------------------- step kernel -------------------------------
// grid = 68 blocks x 256 threads (8 waves).
// Blocks 0..63 : gates for hidden cols [16*nb,16*nb+16), tiles tt = gate 0..3.
// Blocks 64..67: logits for h_{t-1}: [128x1024] x [256x1024]^T, 4 n-tiles each.

__global__ __launch_bounds__(256) void lstm_step(
    const unsigned short* __restrict__ wcat,    // [4096][1280]
    const unsigned short* __restrict__ xbf,     // [512][128][256]
    const float*          __restrict__ biascat, // [4096]
    unsigned short*       __restrict__ hbuf,    // [2][128][1024]
    float*                __restrict__ cst,     // [128][1024]
    const unsigned short* __restrict__ woutbf,  // [256][1024]
    const float*          __restrict__ bout,    // [256]
    float*                __restrict__ logits,  // [512][128][256]
    int t, int do_gates, int do_out)
{
    // 2 buffers x 4 B-tiles of [16][32] bf16 = 8 KB.
    __shared__ alignas(16) unsigned char lds[2 * 4096];

    const int blk  = blockIdx.x;
    const int tid  = threadIdx.x;
    const int lane = tid & 31;
    const int w    = tid >> 5;                         // wave id 0..7
    const int lrow = lane & 15;                        // row within tile
    const int koff = (lane >> 4) * 8;                  // K sub-offset per lane half

    const unsigned short* hprev = hbuf + (size_t)((t + 1) & 1) * BATCH * HID;

    if (blk < 64) {
        if (!do_gates) return;
        unsigned short* hcur = hbuf + (size_t)(t & 1) * BATCH * HID;
        const int nb   = blk;                          // hidden column tile
        const int mrow = 16 * w + lrow;                // batch row for A-frag

        v8f acc[4];
#pragma unroll
        for (int g = 0; g < 4; ++g)
            acc[g] = splat8(biascat[g * HID + nb * 16 + lrow]);

        // B staging: 4 tiles x 16 rows x 4 x 16B chunks = 256 chunks; 1/thread.
        const int stt   = tid >> 6;                    // gate tile 0..3
        const int srow  = (tid >> 2) & 15;
        const int spart = tid & 3;
        const unsigned short* srcrow =
            wcat + (size_t)(stt * HID + nb * 16 + srow) * KTOT + spart * 8;
        unsigned char* dchunk = lds /* + buf*4096 */ + stt * 1024 + srow * 64 + spart * 16;

        // Prologue: stage k-step 0 into buffer 0.
        uint4 stg = *(const uint4*)(srcrow);
        *(uint4*)(dchunk) = stg;

        for (int ks = 0; ks < NKS; ++ks) {
            const int kbase = ks * 32;
            const int cur   = ks & 1;

            // Prefetch next k-step's B chunk into registers (L2-resident).
            uint4 nxt;
            if (ks + 1 < NKS) nxt = *(const uint4*)(srcrow + (ks + 1) * 32);

            // A fragment: h_{t-1} for k<1024, x_t for k>=1024.
            FragU ua;
            const unsigned short* ap;
            if (kbase < HID)
                ap = hprev + (size_t)mrow * HID + kbase + koff;
            else
                ap = xbf + ((size_t)t * BATCH + mrow) * IND + (kbase - HID) + koff;
            ua.q[0] = *(const uint4*)ap;
            ua.q[1] = *(const uint4*)(ap + 16);        // +16 elems = K+16..K+23

            __syncthreads();                           // buf[cur] stores visible

            const unsigned char* base = lds + cur * 4096;
#pragma unroll
            for (int g = 0; g < 4; ++g) {
                FragU ub;
                const unsigned char* bp = base + g * 1024 + lrow * 64 + koff * 2;
                ub.q[0] = *(const uint4*)bp;
                ub.q[1] = *(const uint4*)(bp + 32);
                acc[g] = __builtin_amdgcn_wmma_f32_16x16x32_bf16(
                    false, ua.v, false, ub.v, (short)0, acc[g], false, false);
            }

            // Store next stage into the other buffer (no conflict with readers).
            if (ks + 1 < NKS)
                *(uint4*)(dchunk + ((ks + 1) & 1) * 4096) = nxt;
        }

        // Fused LSTM cell: i/f/g/o for a (b,n) element sit in the same
        // lane/slot across the 4 gate fragments -> pure per-lane VALU.
        const int nhalf = (lane >= 16) ? 8 : 0;
        const int n = nb * 16 + lrow;
#pragma unroll
        for (int r = 0; r < 8; ++r) {
            int brow = 16 * w + r + nhalf;
            float iv = acc[0][r];
            float fv = acc[1][r];
            float gv = acc[2][r];
            float ov = acc[3][r];
            float si = 1.0f / (1.0f + __expf(-iv));
            float sf = 1.0f / (1.0f + __expf(-fv));
            float so = 1.0f / (1.0f + __expf(-ov));
            float tg = tanhf(gv);
            size_t ci = (size_t)brow * HID + n;
            float cn = sf * cst[ci] + si * tg;
            cst[ci]  = cn;
            hcur[ci] = f2bf(so * tanhf(cn));
        }
    } else {
        if (!do_out) return;                           // t==0: no h_{t-1} yet
        const int nb2  = blk - 64;                     // 0..3 -> 4 n-tiles each
        const int mrow = 16 * w + lrow;

        v8f acc[4];
#pragma unroll
        for (int tt = 0; tt < 4; ++tt)
            acc[tt] = splat8(bout[(nb2 * 4 + tt) * 16 + lrow]);

        for (int ks = 0; ks < HID / 32; ++ks) {
            const int kbase = ks * 32;
            FragU ua;
            const unsigned short* ap = hprev + (size_t)mrow * HID + kbase + koff;
            ua.q[0] = *(const uint4*)ap;
            ua.q[1] = *(const uint4*)(ap + 16);
#pragma unroll
            for (int tt = 0; tt < 4; ++tt) {
                FragU ub;
                const unsigned short* bp =
                    woutbf + (size_t)((nb2 * 4 + tt) * 16 + lrow) * HID + kbase + koff;
                ub.q[0] = *(const uint4*)bp;
                ub.q[1] = *(const uint4*)(bp + 16);
                acc[tt] = __builtin_amdgcn_wmma_f32_16x16x32_bf16(
                    false, ua.v, false, ub.v, (short)0, acc[tt], false, false);
            }
        }

        const int nhalf = (lane >= 16) ? 8 : 0;
#pragma unroll
        for (int tt = 0; tt < 4; ++tt) {
            int col = (nb2 * 4 + tt) * 16 + lrow;
#pragma unroll
            for (int r = 0; r < 8; ++r) {
                int brow = 16 * w + r + nhalf;
                logits[((size_t)(t - 1) * BATCH + brow) * IND + col] = acc[tt][r];
            }
        }
    }
}

// ------------------------------ log_softmax --------------------------------
// One wave per row of 256; lane handles 8 strided elements.

__global__ __launch_bounds__(256) void k_logsoftmax(float* __restrict__ out) {
    int wv   = threadIdx.x >> 5;
    int lane = threadIdx.x & 31;
    size_t row = (size_t)blockIdx.x * 8 + wv;          // 65536 rows
    float* p = out + row * IND;

    float x[8];
    float m = -3.4e38f;
#pragma unroll
    for (int i = 0; i < 8; ++i) { x[i] = p[lane + 32 * i]; m = fmaxf(m, x[i]); }
#pragma unroll
    for (int off = 16; off >= 1; off >>= 1) m = fmaxf(m, __shfl_xor(m, off, 32));
    float s = 0.0f;
#pragma unroll
    for (int i = 0; i < 8; ++i) s += __expf(x[i] - m);
#pragma unroll
    for (int off = 16; off >= 1; off >>= 1) s += __shfl_xor(s, off, 32);
    float lse = m + logf(s);
#pragma unroll
    for (int i = 0; i < 8; ++i) p[lane + 32 * i] = x[i] - lse;
}

// -------------------------------- launcher ---------------------------------

extern "C" void kernel_launch(void* const* d_in, const int* in_sizes, int n_in,
                              void* d_out, int out_size, void* d_ws, size_t ws_size,
                              hipStream_t stream) {
    const float* inp  = (const float*)d_in[0];   // [512][128][256]
    const float* Wxh  = (const float*)d_in[1];   // [4096][256]
    const float* bxh  = (const float*)d_in[2];   // [4096]
    const float* Whh  = (const float*)d_in[3];   // [4096][1024]
    const float* bhh  = (const float*)d_in[4];   // [4096]
    const float* Wout = (const float*)d_in[5];   // [256][1024]
    const float* bout = (const float*)d_in[6];   // [256]
    float* out = (float*)d_out;                  // [512][128][256]

    // Workspace layout (~46 MB total), 256B aligned slabs.
    unsigned char* ws = (unsigned char*)d_ws;
    size_t off = 0;
    auto take = [&](size_t bytes) -> unsigned char* {
        unsigned char* p = ws + off;
        off = (off + bytes + 255) & ~(size_t)255;
        return p;
    };
    unsigned short* wcat    = (unsigned short*)take((size_t)G4 * KTOT * 2);        // 10 MB
    unsigned short* xbf     = (unsigned short*)take((size_t)SEQ * BATCH * IND * 2);// 32 MB
    unsigned short* woutbf  = (unsigned short*)take((size_t)IND * HID * 2);        // 0.5 MB
    float*          biascat = (float*)take((size_t)G4 * 4);
    float*          cst     = (float*)take((size_t)BATCH * HID * 4);               // 0.5 MB
    unsigned short* hbuf    = (unsigned short*)take((size_t)2 * BATCH * HID * 2);  // 0.5 MB
    (void)ws_size; (void)in_sizes; (void)n_in; (void)out_size;

    // Prep: casts, fused bias, zero state.
    k_cast_wcat<<<(G4 * KTOT + 255) / 256, 256, 0, stream>>>(Whh, Wxh, wcat);
    k_cast_f32_bf16<<<(SEQ * BATCH * IND + 255) / 256, 256, 0, stream>>>(
        inp, xbf, SEQ * BATCH * IND);
    k_cast_f32_bf16<<<(IND * HID + 255) / 256, 256, 0, stream>>>(
        Wout, woutbf, IND * HID);
    k_bias_cat<<<(G4 + 255) / 256, 256, 0, stream>>>(bxh, bhh, biascat);
    k_init_state<<<(2 * BATCH * HID + 255) / 256, 256, 0, stream>>>(cst, hbuf);

    // Serial scan: step t computes h_t (blocks 0..63) and logits for t-1
    // (blocks 64..67). t==SEQ is the tail that emits logits for t=511.
    for (int t = 0; t <= SEQ; ++t) {
        int dg  = (t < SEQ) ? 1 : 0;
        int doo = (t >= 1) ? 1 : 0;
        lstm_step<<<68, 256, 0, stream>>>(wcat, xbf, biascat, hbuf, cst,
                                          woutbf, bout, out, t, dg, doo);
    }

    // In-place log_softmax over 65536 rows of 256.
    k_logsoftmax<<<(SEQ * BATCH) / 8, 256, 0, stream>>>(out);
}
